// Trilinear_9491877724842
// MI455X (gfx1250) — matrix operand order repaired and checked by reading, not compiled, not run on hardware
//
#include <hip/hip_runtime.h>

// MI455X / gfx1250 trilinear contraction:
//   y[b,a] = sum_{n,m,o} x1[b,n] x2[b,m] x3[b,o] W[a,n,m,o] + bias[a]
// Factored as GEMM  [B x 4096 (P = x2 (x) x3)] x [4096 x 8192 (W rows a*64+n)]
// with the n-contraction against x1 fused into the epilogue.
// f16 WMMA (V_WMMA_F32_16X16X32_F16), fp32 accumulate.
// Double-buffered LDS with GLOBAL_LOAD_ASYNC_TO_LDS_B128 prefetch (ASYNCcnt).

typedef _Float16 h16;
typedef __attribute__((ext_vector_type(16))) _Float16 v16h;
typedef __attribute__((ext_vector_type(8)))  _Float16 v8h;
typedef __attribute__((ext_vector_type(8)))  float    v8f;
typedef __attribute__((ext_vector_type(4)))  float    v4f;

namespace {
constexpr int kB  = 16384;     // batch
constexpr int kN  = 64;        // in1 dim
constexpr int kM  = 64;        // in2 dim
constexpr int kO  = 64;        // in3 dim
constexpr int kA  = 128;       // out dim
constexpr int kK  = kM * kO;   // 4096 GEMM K
constexpr int BT  = 64;        // batch rows / workgroup (4 WMMA M-tiles)
constexpr int AG  = 8;         // a-values / workgroup (one per wave)
constexpr int CT  = AG * kN;   // 512 GEMM columns / workgroup
constexpr int KT  = 32;        // K step = one f16 WMMA depth
constexpr int LDK = 40;        // padded LDS k-stride: 80B rows, 16B aligned
constexpr int NSTEP = kK / KT; // 128
}

union V16U { v16h v; v8h h[2]; };

__device__ __forceinline__ v8h cvt8(v4f a, v4f b) {
  v8h o;
  o[0] = (h16)a[0]; o[1] = (h16)a[1]; o[2] = (h16)a[2]; o[3] = (h16)a[3];
  o[4] = (h16)b[0]; o[5] = (h16)b[1]; o[6] = (h16)b[2]; o[7] = (h16)b[3];
  return o;
}

// ---- Kernel 1: W fp32 -> f16 (one-time; ~200MB HBM ~= 9us at 23.3 TB/s)
__global__ __launch_bounds__(256) void wcvt_f16(const float* __restrict__ w,
                                                h16* __restrict__ o, long n) {
  long i = ((long)blockIdx.x * 256 + (long)threadIdx.x) * 8;
  if (i + 8 > n) return;
  v4f f0 = *(const v4f*)(w + i);
  v4f f1 = *(const v4f*)(w + i + 4);
  *(v8h*)(o + i) = cvt8(f0, f1);
}

// ---- Kernel 2: fused outer-product -> WMMA GEMM -> x1 reduction epilogue
template <bool PRECVT>
__global__ __launch_bounds__(256) void trilinear_wmma_9491(
    const float* __restrict__ x1, const float* __restrict__ x2,
    const float* __restrict__ x3, const float* __restrict__ wf,
    const h16* __restrict__ wh, const float* __restrict__ bias,
    float* __restrict__ out) {
  __shared__ __align__(16) h16 As[2][BT][LDK];  // P tiles (64 x 32 used)
  __shared__ __align__(16) h16 Bs[2][CT][LDK];  // W tiles, k-major per column
  __shared__ float X1[BT][kN];                  // x1 tile for epilogue

  const int t  = threadIdx.x;
  const int R0 = blockIdx.x * BT;  // batch row base
  const int G0 = blockIdx.y * CT;  // W row base = a_base*64

  // Stage x1[R0..R0+63, 0..63] (contiguous, coalesced)
#pragma unroll
  for (int i = 0; i < (BT * kN) / 256; ++i)
    ((float*)X1)[t + 256 * i] = x1[(size_t)R0 * kN + t + 256 * i];

  const int lane = t & 31;
  const int wv   = t >> 5;   // wave id 0..7 -> which 'a'
  const int li   = lane & 15;
  const int half = lane >> 4;
  const int cw   = t >> 2;   // this thread's column base (0..63)
  const int ch   = t & 3;    // 16B k-chunk id (0..3)

  // Async-copy addressing: per-thread chunk j covers column cw + 64*j.
  const h16* wbase =
      PRECVT ? (wh + (size_t)G0 * kK) : (const h16*)nullptr;
  const unsigned ldsB =
      (unsigned)(uintptr_t)&Bs[0][0][0] + (unsigned)(cw * (LDK * 2) + ch * 16);
  const unsigned vof      = (unsigned)(cw * (kK * 2) + ch * 16);
  constexpr unsigned BUFB = (unsigned)(CT * LDK * 2);  // bytes per Bs buffer
  constexpr unsigned JSL  = 64u * LDK * 2;             // lds stride per chunk j
  constexpr unsigned JSG  = 64u * kK * 2;              // glb stride per chunk j

  auto issueB = [&](int buf, int K0) {
    if constexpr (PRECVT) {
      const unsigned lb = ldsB + (unsigned)buf * BUFB;
      const unsigned gb = vof + (unsigned)(K0 * 2);
#pragma unroll
      for (int j = 0; j < 8; ++j) {
        unsigned l = lb + (unsigned)j * JSL;
        unsigned g = gb + (unsigned)j * JSG;
        // ASYNCcnt-tracked DMA into LDS; bypasses VGPRs entirely.
        asm volatile("global_load_async_to_lds_b128 %0, %1, %2"
                     :: "v"(l), "v"(g), "s"(wbase)
                     : "memory");
      }
    } else {
#pragma unroll
      for (int j = 0; j < 8; ++j) {
        const float* p =
            wf + (size_t)(G0 + cw + 64 * j) * kK + K0 + ch * 8;
        *(v8h*)&Bs[buf][cw + 64 * j][ch * 8] =
            cvt8(*(const v4f*)p, *(const v4f*)(p + 4));
      }
    }
  };

  // P tile: P[b,k] = x2[b,m] * x3[b,o]; m fixed within a K-step.
  auto buildA = [&](int buf, int K0) {
    const int   m  = K0 >> 6;
    const int   o0 = K0 & 63;  // 0 or 32
    const float xv = x2[(size_t)(R0 + cw) * kM + m];
    const float* p3 = x3 + (size_t)(R0 + cw) * kO + o0 + ch * 8;
    v4f f0 = *(const v4f*)p3;
    v4f f1 = *(const v4f*)(p3 + 4);
    f0 *= xv;
    f1 *= xv;
    *(v8h*)&As[buf][cw][ch * 8] = cvt8(f0, f1);
  };

  v8f acc[4][4] = {};  // 4 M-tiles x 4 N-tiles per wave

  issueB(0, 0);
  buildA(0, 0);

  for (int i = 0; i < NSTEP; ++i) {
    const int cur = i & 1;
    const int nxt = cur ^ 1;
    if (i) __syncthreads();  // all reads of buffer[nxt] (iter i-1) are done
    if (i + 1 < NSTEP) {     // prefetch next K-step while computing this one
      issueB(nxt, (i + 1) * KT);
      buildA(nxt, (i + 1) * KT);
    }
    if constexpr (PRECVT) {
      // In-order async completion: leaving <=8 outstanding retires cur's 8.
      if (i + 1 < NSTEP)
        asm volatile("s_wait_asynccnt 0x8" ::: "memory");
      else
        asm volatile("s_wait_asynccnt 0x0" ::: "memory");
    }
    __syncthreads();  // buffer[cur] (async + ds_store) visible to all waves

    // Fragment loads per ISA 16-bit A/B layouts (half-wave K split).
    V16U af[4], bfr[4];
#pragma unroll
    for (int mt = 0; mt < 4; ++mt) {
      const int r = mt * 16 + li;
      af[mt].h[0] = *(const v8h*)&As[cur][r][half * 8];       // K 0..7 / 8..15
      af[mt].h[1] = *(const v8h*)&As[cur][r][16 + half * 8];  // K 16..23/24..31
    }
#pragma unroll
    for (int nt = 0; nt < 4; ++nt) {
      const int c = wv * 64 + nt * 16 + li;
      bfr[nt].h[0] = *(const v8h*)&Bs[cur][c][half * 8];       // K 0..7 / 8..15
      bfr[nt].h[1] = *(const v8h*)&Bs[cur][c][16 + half * 8];  // K 16..23/24..31
    }
#pragma unroll
    for (int mt = 0; mt < 4; ++mt)
#pragma unroll
      for (int nt = 0; nt < 4; ++nt)
        acc[mt][nt] = __builtin_amdgcn_wmma_f32_16x16x32_f16(
            false, af[mt].v, false, bfr[nt].v, (short)0, acc[mt][nt], false,
            false);
  }

  // Epilogue: y[b,a] = sum_n x1[b,n] * t[b,a,n] + bias[a]
  // Lane li holds column n = nt*16+li; VGPR v of acc = row mt*16+half*8+v.
  const int   a  = (G0 >> 6) + wv;
  const float bv = bias[a];
#pragma unroll
  for (int mt = 0; mt < 4; ++mt) {
#pragma unroll
    for (int v = 0; v < 8; ++v) {
      const int rl = mt * 16 + half * 8 + v;
      float s = 0.f;
#pragma unroll
      for (int nt = 0; nt < 4; ++nt)
        s += acc[mt][nt][v] * X1[rl][nt * 16 + li];
      // XOR butterfly over the 16 n-lanes of this half-wave (ds_swizzle).
      s += __int_as_float(
          __builtin_amdgcn_ds_swizzle(__float_as_int(s), 0x041F));  // xor 1
      s += __int_as_float(
          __builtin_amdgcn_ds_swizzle(__float_as_int(s), 0x081F));  // xor 2
      s += __int_as_float(
          __builtin_amdgcn_ds_swizzle(__float_as_int(s), 0x101F));  // xor 4
      s += __int_as_float(
          __builtin_amdgcn_ds_swizzle(__float_as_int(s), 0x201F));  // xor 8
      if (li == 0) out[(size_t)(R0 + rl) * kA + a] = s + bv;
    }
  }
}

extern "C" void kernel_launch(void* const* d_in, const int* in_sizes, int n_in,
                              void* d_out, int out_size, void* d_ws,
                              size_t ws_size, hipStream_t stream) {
  const float* x1   = (const float*)d_in[0];
  const float* x2   = (const float*)d_in[1];
  const float* x3   = (const float*)d_in[2];
  const float* w    = (const float*)d_in[3];
  const float* bias = (const float*)d_in[4];
  float*       out  = (float*)d_out;

  const size_t wElems = (size_t)kA * kN * kK;  // 33,554,432
  dim3 grid(kB / BT, kA / AG);                 // (256, 16)

  if (ws_size >= wElems * sizeof(h16)) {
    h16* wh = (h16*)d_ws;
    wcvt_f16<<<(int)(wElems / 2048), 256, 0, stream>>>(w, wh, (long)wElems);
    trilinear_wmma_9491<true><<<grid, 256, 0, stream>>>(x1, x2, x3, nullptr,
                                                        wh, bias, out);
  } else {
    // Workspace too small: convert W fp32->f16 inline while tiling (2x L2
    // traffic for W, still correct and WMMA-driven).
    trilinear_wmma_9491<false><<<grid, 256, 0, stream>>>(x1, x2, x3, w,
                                                         nullptr, bias, out);
  }
}
